// AdvancedLLM_22668837388555
// MI455X (gfx1250) — compile-verified, matrix operand order checked by reading, not run on hardware
//
#include <hip/hip_runtime.h>
#include <hip/hip_bf16.h>

// ---------------- problem constants ----------------
constexpr int B_   = 2;
constexpr int S_   = 2048;
constexpr int D_   = 1024;
constexpr int NH_  = 16;
constexpr int G_   = 4;
constexpr int DK_  = 64;      // D_/NH_
constexpr int HPG_ = 4;       // NH_/G_
constexpr int FF_  = 4096;
constexpr int E_   = 8;
constexpr float EPS_ = 1e-6f;
constexpr int BS_  = B_ * S_; // 4096 tokens

typedef __bf16 bf16;
typedef __bf16 bf16x16 __attribute__((ext_vector_type(16)));
typedef float  f32x8   __attribute__((ext_vector_type(8)));

__device__ __forceinline__ f32x8 fzero() {
  f32x8 z = {0.f,0.f,0.f,0.f,0.f,0.f,0.f,0.f};
  return z;
}

__device__ __forceinline__ f32x8 wmma_bf16(bf16x16 a, bf16x16 b, f32x8 c) {
  // (neg_a, A, neg_b, B, c_mod, C, reuse_a, reuse_b)
  return __builtin_amdgcn_wmma_f32_16x16x32_bf16(false, a, false, b, (short)0, c, false, false);
}

// Load a 16x32 bf16 fragment from a row-major tile (A-layout; B uses the same
// pattern because weights are stored pre-transposed, N-major).
__device__ __forceinline__ bf16x16 load_frag(const bf16* base, int ld, int lane) {
  int r  = lane & 15;
  int kc = (lane & 16) ? 8 : 0;
  const bf16* p = base + r * ld + kc;
  bf16x16 f;
  *(uint4*)&f       = *(const uint4*)(p);
  *((uint4*)&f + 1) = *(const uint4*)(p + 16);
  return f;
}

// CDNA5 async global->LDS copy, 16 bytes per lane, tracked by ASYNCcnt.
__device__ __forceinline__ void async_copy_b128(void* lds_dst, const void* gsrc) {
  unsigned d = (unsigned)(size_t)lds_dst;              // LDS byte offset (low 32 bits)
  unsigned long long s = (unsigned long long)(size_t)gsrc;
  asm volatile("global_load_async_to_lds_b128 %0, %1, off" :: "v"(d), "v"(s) : "memory");
}
#define ASYNC_WAIT(n) asm volatile("s_wait_asynccnt " #n ::: "memory")

// ---------------- fp32 -> bf16 convert + transpose (weights) ----------------
__global__ void cvt_transpose(const float* __restrict__ in, bf16* __restrict__ out,
                              int K, int N, long strideIn, long strideOut) {
  const float* ip = in + (long)blockIdx.z * strideIn;
  bf16* op = out + (long)blockIdx.z * strideOut;
  __shared__ bf16 tile[32][33];
  const int k0 = blockIdx.y * 32, n0 = blockIdx.x * 32;
  for (int i = threadIdx.y; i < 32; i += 8)
    tile[i][threadIdx.x] = (bf16)ip[(long)(k0 + i) * N + n0 + threadIdx.x];
  __syncthreads();
  for (int i = threadIdx.y; i < 32; i += 8)
    op[(long)(n0 + i) * K + k0 + threadIdx.x] = tile[threadIdx.x][i];
}

// ---------------- RMSNorm + RoPE (token per block) ----------------
__global__ void rmsnorm_rope_kernel(const float* __restrict__ x,
                                    const float* __restrict__ w,
                                    bf16* __restrict__ xr) {
  const int t = blockIdx.x;          // token
  const int s = t % S_;              // position
  const float* xt = x + (long)t * D_;
  __shared__ float red[256];
  float ss = 0.f;
  for (int d = threadIdx.x; d < D_; d += 256) { float v = xt[d]; ss += v * v; }
  red[threadIdx.x] = ss;
  __syncthreads();
  for (int off = 128; off > 0; off >>= 1) {
    if (threadIdx.x < off) red[threadIdx.x] += red[threadIdx.x + off];
    __syncthreads();
  }
  const float rinv = rsqrtf(red[0] / (float)D_ + EPS_);
  bf16* ot = xr + (long)t * D_;
  const int half = D_ / 2;
  for (int j = threadIdx.x; j < half; j += 256) {
    float xe = xt[2*j]   * rinv * w[2*j];
    float xo = xt[2*j+1] * rinv * w[2*j+1];
    float theta = __powf(10000.0f, -(float)j / (float)half);
    float c, sn;
    __sincosf((float)s * theta, &c, &sn);
    ot[j]        = (bf16)(xe * c - xo * sn);
    ot[j + half] = (bf16)(xe * sn + xo * c);
  }
}

// ---------------- plain RMSNorm (token per block) ----------------
__global__ void rmsnorm_kernel(const float* __restrict__ x,
                               const float* __restrict__ w,
                               bf16* __restrict__ y) {
  const int t = blockIdx.x;
  const float* xt = x + (long)t * D_;
  __shared__ float red[256];
  float ss = 0.f;
  for (int d = threadIdx.x; d < D_; d += 256) { float v = xt[d]; ss += v * v; }
  red[threadIdx.x] = ss;
  __syncthreads();
  for (int off = 128; off > 0; off >>= 1) {
    if (threadIdx.x < off) red[threadIdx.x] += red[threadIdx.x + off];
    __syncthreads();
  }
  const float rinv = rsqrtf(red[0] / (float)D_ + EPS_);
  bf16* ot = y + (long)t * D_;
  for (int d = threadIdx.x; d < D_; d += 256) ot[d] = (bf16)(xt[d] * rinv * w[d]);
}

// ---------------- 64x128 GEMM core, double-buffered async pipeline ----------------
// A: (M x K) row-major activations. Bt: (N x K) row-major (pre-transposed weights).
// 128 threads = 4 waves arranged 2(m) x 2(n); each wave computes 32x64 (2x4 WMMA).
// Per thread per tile: 2 A copies + 4 B copies = 6 async ops -> ASYNC_WAIT(6).
__device__ __forceinline__ void gemm_issue(const bf16* __restrict__ A,
                                           const bf16* __restrict__ Bt,
                                           int K, int m0, int n0, int k0,
                                           bf16* ldsA, bf16* ldsB, int tid) {
  #pragma unroll
  for (int i = 0; i < 2; ++i) {
    int c = tid + i * 128;                 // 256 chunks of 8
    int r = c >> 2, c8 = (c & 3) * 8;
    async_copy_b128(&ldsA[r * 32 + c8], &A[(long)(m0 + r) * K + k0 + c8]);
  }
  #pragma unroll
  for (int i = 0; i < 4; ++i) {
    int c = tid + i * 128;                 // 512 chunks of 8
    int r = c >> 2, c8 = (c & 3) * 8;
    async_copy_b128(&ldsB[r * 32 + c8], &Bt[(long)(n0 + r) * K + k0 + c8]);
  }
}

__device__ __forceinline__ void gemm_compute(const bf16* curA, const bf16* curB,
                                             int wm, int wn, int lane, f32x8 acc[2][4]) {
  bf16x16 a0 = load_frag(curA + (wm +  0) * 32, 32, lane);
  bf16x16 a1 = load_frag(curA + (wm + 16) * 32, 32, lane);
  #pragma unroll
  for (int j = 0; j < 4; ++j) {
    bf16x16 b = load_frag(curB + (wn + j * 16) * 32, 32, lane);
    acc[0][j] = wmma_bf16(a0, b, acc[0][j]);
    acc[1][j] = wmma_bf16(a1, b, acc[1][j]);
  }
}

__device__ void gemm_core(const bf16* __restrict__ A, const bf16* __restrict__ Bt,
                          int K, int m0, int n0,
                          bf16* ldsA, bf16* ldsB, f32x8 acc[2][4]) {
  const int tid = threadIdx.x, lane = tid & 31, w = tid >> 5;
  const int wm = (w >> 1) * 32, wn = (w & 1) * 64;
  const int ntile = K >> 5;
  gemm_issue(A, Bt, K, m0, n0, 0, ldsA, ldsB, tid);
  int t = 0;
  // steady state: issue next tile, drain current, compute (branch-free wait)
  for (; t < ntile - 1; ++t) {
    gemm_issue(A, Bt, K, m0, n0, (t + 1) << 5,
               ldsA + ((t + 1) & 1) * (64 * 32),
               ldsB + ((t + 1) & 1) * (128 * 32), tid);
    ASYNC_WAIT(6);
    __syncthreads();
    gemm_compute(ldsA + (t & 1) * (64 * 32), ldsB + (t & 1) * (128 * 32),
                 wm, wn, lane, acc);
    __syncthreads();
  }
  // peeled last tile
  ASYNC_WAIT(0);
  __syncthreads();
  gemm_compute(ldsA + (t & 1) * (64 * 32), ldsB + (t & 1) * (128 * 32),
               wm, wn, lane, acc);
}

#define GEMM_PROLOGUE()                                                    \
  __shared__ __align__(16) bf16 ldsA[2 * 64 * 32];                         \
  __shared__ __align__(16) bf16 ldsB[2 * 128 * 32];                        \
  const int m0 = blockIdx.y * 64, n0 = blockIdx.x * 128;                   \
  f32x8 acc[2][4] = {fzero(), fzero(), fzero(), fzero(),                   \
                     fzero(), fzero(), fzero(), fzero()};                  \
  gemm_core(A, Bt, K, m0, n0, ldsA, ldsB, acc);                            \
  const int lane = threadIdx.x & 31, w = threadIdx.x >> 5;                 \
  const int rbase = m0 + (w >> 1) * 32 + ((lane & 16) ? 8 : 0);            \
  const int col0  = n0 + (w & 1) * 64 + (lane & 15);

// C (bf16, row-major M x N) = A @ Bt^T
__global__ void gemm_bf16_out(const bf16* __restrict__ A, const bf16* __restrict__ Bt,
                              bf16* __restrict__ C, int N, int K) {
  GEMM_PROLOGUE();
  #pragma unroll
  for (int si = 0; si < 2; ++si)
    #pragma unroll
    for (int sj = 0; sj < 4; ++sj)
      #pragma unroll
      for (int i = 0; i < 8; ++i)
        C[(long)(rbase + si * 16 + i) * N + col0 + sj * 16] = (bf16)acc[si][sj][i];
}

// V projection with transposed store: Vt[(b*G*DK + col) * S + s] (dk-major for attention)
__global__ void gemm_bf16_outT(const bf16* __restrict__ A, const bf16* __restrict__ Bt,
                               bf16* __restrict__ Vt, int N, int K) {
  GEMM_PROLOGUE();
  #pragma unroll
  for (int si = 0; si < 2; ++si)
    #pragma unroll
    for (int sj = 0; sj < 4; ++sj)
      #pragma unroll
      for (int i = 0; i < 8; ++i) {
        int row = rbase + si * 16 + i;           // token
        int bb  = row >> 11;                     // /S_
        int s   = row & (S_ - 1);
        int col = col0 + sj * 16;                // g*DK + dk
        Vt[((long)bb * (G_ * DK_) + col) * S_ + s] = (bf16)acc[si][sj][i];
      }
}

// C (f32) = A @ Bt^T + resid  (Wo projection + residual)
__global__ void gemm_wo_resid(const bf16* __restrict__ A, const bf16* __restrict__ Bt,
                              const float* __restrict__ resid, float* __restrict__ C,
                              int N, int K) {
  GEMM_PROLOGUE();
  #pragma unroll
  for (int si = 0; si < 2; ++si)
    #pragma unroll
    for (int sj = 0; sj < 4; ++sj)
      #pragma unroll
      for (int i = 0; i < 8; ++i) {
        long idx = (long)(rbase + si * 16 + i) * N + col0 + sj * 16;
        C[idx] = acc[si][sj][i] + resid[idx];
      }
}

// Out (f32) += gate[row] * (A @ Bt^T + bias)  (MoE down-proj, gated accumulate)
__global__ void gemm_w3_gate(const bf16* __restrict__ A, const bf16* __restrict__ Bt,
                             const float* __restrict__ bias, const float* __restrict__ gate,
                             float* __restrict__ Out, int N, int K) {
  GEMM_PROLOGUE();
  #pragma unroll
  for (int si = 0; si < 2; ++si)
    #pragma unroll
    for (int sj = 0; sj < 4; ++sj)
      #pragma unroll
      for (int i = 0; i < 8; ++i) {
        int row = rbase + si * 16 + i;
        int col = col0 + sj * 16;
        float g = gate[(long)row * E_];
        Out[(long)row * N + col] += g * (acc[si][sj][i] + bias[col]);
      }
}

// H (bf16) = silu(A@B1t^T + b1) * (A@B2t^T + b2) — 64x64 tile, dual accumulators,
// double-buffered (6 async ops per thread per tile)
__device__ __forceinline__ void dual_issue(const bf16* __restrict__ A,
                                           const bf16* __restrict__ B1t,
                                           const bf16* __restrict__ B2t,
                                           int K, int m0, int n0, int k0,
                                           bf16* ldsA, bf16* ldsB1, bf16* ldsB2, int tid) {
  #pragma unroll
  for (int i = 0; i < 2; ++i) {
    int c = tid + i * 128;
    int r = c >> 2, c8 = (c & 3) * 8;
    async_copy_b128(&ldsA[r * 32 + c8],  &A[(long)(m0 + r) * K + k0 + c8]);
    async_copy_b128(&ldsB1[r * 32 + c8], &B1t[(long)(n0 + r) * K + k0 + c8]);
    async_copy_b128(&ldsB2[r * 32 + c8], &B2t[(long)(n0 + r) * K + k0 + c8]);
  }
}

__device__ __forceinline__ void dual_compute(const bf16* cA, const bf16* cB1, const bf16* cB2,
                                             int wm, int wn, int lane,
                                             f32x8 acc1[2][2], f32x8 acc2[2][2]) {
  bf16x16 a0 = load_frag(cA + (wm +  0) * 32, 32, lane);
  bf16x16 a1 = load_frag(cA + (wm + 16) * 32, 32, lane);
  bf16x16 p0 = load_frag(cB1 + (wn +  0) * 32, 32, lane);
  bf16x16 p1 = load_frag(cB1 + (wn + 16) * 32, 32, lane);
  bf16x16 q0 = load_frag(cB2 + (wn +  0) * 32, 32, lane);
  bf16x16 q1 = load_frag(cB2 + (wn + 16) * 32, 32, lane);
  acc1[0][0] = wmma_bf16(a0, p0, acc1[0][0]);
  acc1[0][1] = wmma_bf16(a0, p1, acc1[0][1]);
  acc1[1][0] = wmma_bf16(a1, p0, acc1[1][0]);
  acc1[1][1] = wmma_bf16(a1, p1, acc1[1][1]);
  acc2[0][0] = wmma_bf16(a0, q0, acc2[0][0]);
  acc2[0][1] = wmma_bf16(a0, q1, acc2[0][1]);
  acc2[1][0] = wmma_bf16(a1, q0, acc2[1][0]);
  acc2[1][1] = wmma_bf16(a1, q1, acc2[1][1]);
}

__global__ void gemm_dual_silu(const bf16* __restrict__ A,
                               const bf16* __restrict__ B1t, const bf16* __restrict__ B2t,
                               const float* __restrict__ bias1, const float* __restrict__ bias2,
                               bf16* __restrict__ H, int N, int K) {
  __shared__ __align__(16) bf16 ldsA[2 * 64 * 32];
  __shared__ __align__(16) bf16 ldsB1[2 * 64 * 32];
  __shared__ __align__(16) bf16 ldsB2[2 * 64 * 32];
  const int m0 = blockIdx.y * 64, n0 = blockIdx.x * 64;
  const int tid = threadIdx.x, lane = tid & 31, w = tid >> 5;
  const int wm = (w >> 1) * 32, wn = (w & 1) * 32;
  f32x8 acc1[2][2] = {fzero(), fzero(), fzero(), fzero()};
  f32x8 acc2[2][2] = {fzero(), fzero(), fzero(), fzero()};
  const int ntile = K >> 5;
  dual_issue(A, B1t, B2t, K, m0, n0, 0, ldsA, ldsB1, ldsB2, tid);
  int t = 0;
  for (; t < ntile - 1; ++t) {
    const int nxt = ((t + 1) & 1) * (64 * 32);
    dual_issue(A, B1t, B2t, K, m0, n0, (t + 1) << 5,
               ldsA + nxt, ldsB1 + nxt, ldsB2 + nxt, tid);
    ASYNC_WAIT(6);
    __syncthreads();
    const int cur = (t & 1) * (64 * 32);
    dual_compute(ldsA + cur, ldsB1 + cur, ldsB2 + cur, wm, wn, lane, acc1, acc2);
    __syncthreads();
  }
  ASYNC_WAIT(0);
  __syncthreads();
  {
    const int cur = (t & 1) * (64 * 32);
    dual_compute(ldsA + cur, ldsB1 + cur, ldsB2 + cur, wm, wn, lane, acc1, acc2);
  }
  const int rbase = m0 + wm + ((lane & 16) ? 8 : 0);
  const int col0  = n0 + wn + (lane & 15);
  #pragma unroll
  for (int si = 0; si < 2; ++si)
    #pragma unroll
    for (int sj = 0; sj < 2; ++sj)
      #pragma unroll
      for (int i = 0; i < 8; ++i) {
        int   col = col0 + sj * 16;
        float v1  = acc1[si][sj][i] + bias1[col];
        float v2  = acc2[si][sj][i] + bias2[col];
        float sil = v1 / (1.f + __expf(-v1));
        H[(long)(rbase + si * 16 + i) * N + col] = (bf16)(sil * v2);
      }
}

// ---------------- flash attention (WMMA), causal, GQA ----------------
// grid: (S/64, NH, B), block: 128 (4 waves, 16 queries each)
// K: (b*S + t, G*DK) row-major; Vt: (b*G*DK + dk, S) row-major (pre-transposed)
// Per thread per kv tile: 2 K copies + 2 V copies = 4 async ops -> ASYNC_WAIT(4).
__device__ __forceinline__ void attn_issue(const bf16* __restrict__ K,
                                           const bf16* __restrict__ vbase,
                                           int b, int g, int kv0,
                                           bf16* ldsK, bf16* ldsVt, int tid) {
  #pragma unroll
  for (int i = 0; i < 2; ++i) {
    int c  = tid + i * 128;
    int kr = c >> 3, c8 = (c & 7) * 8;        // K tile: 32 keys x 64 dk
    async_copy_b128(&ldsK[kr * 64 + c8],
                    &K[((long)(b * S_ + kv0 + kr)) * (G_ * DK_) + g * DK_ + c8]);
    int dk = c >> 2, kv8 = (c & 3) * 8;       // V tile: 64 dk x 32 kv
    async_copy_b128(&ldsVt[dk * 32 + kv8], &vbase[(long)dk * S_ + kv0 + kv8]);
  }
}

__global__ void attn_kernel(const bf16* __restrict__ Q, const bf16* __restrict__ K,
                            const bf16* __restrict__ Vt, bf16* __restrict__ O) {
  const int qt = blockIdx.x * 64;
  const int h  = blockIdx.y;
  const int b  = blockIdx.z;
  const int g  = h / HPG_;
  __shared__ __align__(16) bf16 ldsK[2 * 32 * 64];
  __shared__ __align__(16) bf16 ldsVt[2 * 64 * 32];
  __shared__ __align__(16) bf16 ldsP[4][16 * 32];

  const int tid = threadIdx.x, lane = tid & 31, w = tid >> 5;
  const int qrow0 = qt + w * 16;
  const int half8 = (lane & 16) ? 8 : 0;
  const int nlo   = lane & 15;
  const float scale = 0.125f;  // 1/sqrt(DK)

  const bf16* qbase = Q + ((long)(b * S_ + qrow0)) * D_ + h * DK_;
  bf16x16 aq0 = load_frag(qbase, D_, lane);        // dk 0..31
  bf16x16 aq1 = load_frag(qbase + 32, D_, lane);   // dk 32..63

  f32x8 m_i, l_i, o[4];
  #pragma unroll
  for (int i = 0; i < 8; ++i) { m_i[i] = -1e30f; l_i[i] = 0.f; }
  #pragma unroll
  for (int nt = 0; nt < 4; ++nt) o[nt] = fzero();

  const bf16* vbase = Vt + (long)b * (G_ * DK_) * S_ + (long)g * DK_ * S_;

  // one kv step (everything between the tile barriers)
  auto kv_step = [&](int t) {
    const int kv0 = t << 5;
    const bf16* curK = ldsK  + (t & 1) * (32 * 64);
    const bf16* curV = ldsVt + (t & 1) * (64 * 32);

    // scores S = Q @ K^T for 2 key subtiles of 16
    f32x8 sc[2];
    #pragma unroll
    for (int c = 0; c < 2; ++c) {
      bf16x16 bk_lo = load_frag(curK + c * 16 * 64,      64, lane);
      bf16x16 bk_hi = load_frag(curK + c * 16 * 64 + 32, 64, lane);
      sc[c] = wmma_bf16(aq0, bk_lo, fzero());
      sc[c] = wmma_bf16(aq1, bk_hi, sc[c]);
      #pragma unroll
      for (int i = 0; i < 8; ++i) {
        int kj = kv0 + c * 16 + nlo;
        int qi = qrow0 + half8 + i;
        float v = sc[c][i] * scale;
        sc[c][i] = (kj > qi) ? -1e30f : v;
      }
    }

    // online softmax (row stats reduced across 16-lane halves)
    f32x8 rmax;
    #pragma unroll
    for (int i = 0; i < 8; ++i) rmax[i] = fmaxf(sc[0][i], sc[1][i]);
    for (int msk = 1; msk < 16; msk <<= 1)
      #pragma unroll
      for (int i = 0; i < 8; ++i)
        rmax[i] = fmaxf(rmax[i], __shfl_xor(rmax[i], msk, 32));

    f32x8 mnew, alpha, rsum;
    #pragma unroll
    for (int i = 0; i < 8; ++i) {
      mnew[i]  = fmaxf(m_i[i], rmax[i]);
      alpha[i] = __expf(m_i[i] - mnew[i]);
      rsum[i]  = 0.f;
    }
    #pragma unroll
    for (int c = 0; c < 2; ++c)
      #pragma unroll
      for (int i = 0; i < 8; ++i) {
        float p = __expf(sc[c][i] - mnew[i]);
        sc[c][i] = p;
        rsum[i] += p;
      }
    for (int msk = 1; msk < 16; msk <<= 1)
      #pragma unroll
      for (int i = 0; i < 8; ++i)
        rsum[i] += __shfl_xor(rsum[i], msk, 32);
    #pragma unroll
    for (int i = 0; i < 8; ++i) {
      l_i[i] = l_i[i] * alpha[i] + rsum[i];
      m_i[i] = mnew[i];
    }
    #pragma unroll
    for (int nt = 0; nt < 4; ++nt)
      #pragma unroll
      for (int i = 0; i < 8; ++i) o[nt][i] *= alpha[i];

    // write P (16x32) to this wave's LDS tile, re-load as A fragment
    bf16* myP = &ldsP[w][0];
    #pragma unroll
    for (int c = 0; c < 2; ++c)
      #pragma unroll
      for (int i = 0; i < 8; ++i)
        myP[(half8 + i) * 32 + c * 16 + nlo] = (bf16)sc[c][i];
    asm volatile("s_wait_dscnt 0x0" ::: "memory");  // per-wave LDS RAW
    bf16x16 aP = load_frag(myP, 32, lane);

    // O += P @ V  (kv-dim = 32, four dk subtiles of 16)
    #pragma unroll
    for (int nt = 0; nt < 4; ++nt) {
      bf16x16 bv = load_frag(curV + nt * 16 * 32, 32, lane);
      o[nt] = wmma_bf16(aP, bv, o[nt]);
    }
  };

  const int ntile = (qt + 64) >> 5;   // causal upper bound / 32
  attn_issue(K, vbase, b, g, 0, ldsK, ldsVt, tid);
  int t = 0;
  for (; t < ntile - 1; ++t) {
    attn_issue(K, vbase, b, g, (t + 1) << 5,
               ldsK + ((t + 1) & 1) * (32 * 64),
               ldsVt + ((t + 1) & 1) * (64 * 32), tid);
    ASYNC_WAIT(4);
    __syncthreads();
    kv_step(t);
    __syncthreads();
  }
  ASYNC_WAIT(0);
  __syncthreads();
  kv_step(t);

  // epilogue: normalize and write to (b, s, h*DK + dk)
  #pragma unroll
  for (int nt = 0; nt < 4; ++nt)
    #pragma unroll
    for (int i = 0; i < 8; ++i) {
      int row = qrow0 + half8 + i;
      int col = h * DK_ + nt * 16 + nlo;
      O[((long)(b * S_ + row)) * D_ + col] = (bf16)(o[nt][i] / l_i[i]);
    }
}

// ---------------- router: logits + top-2 softmax gates ----------------
__global__ void router_gates(const bf16* __restrict__ yn, const float* __restrict__ rw,
                             const float* __restrict__ rb, float* __restrict__ gates) {
  int t = blockIdx.x * blockDim.x + threadIdx.x;
  if (t >= BS_) return;
  float acc[E_];
  #pragma unroll
  for (int e = 0; e < E_; ++e) acc[e] = rb[e];
  const bf16* y = yn + (long)t * D_;
  for (int d = 0; d < D_; ++d) {
    float yv = (float)y[d];
    #pragma unroll
    for (int e = 0; e < E_; ++e) acc[e] += yv * rw[d * E_ + e];
  }
  int i1 = 0; float l1 = acc[0];
  #pragma unroll
  for (int e = 1; e < E_; ++e) if (acc[e] > l1) { l1 = acc[e]; i1 = e; }
  int i2 = -1; float l2 = -3.0e38f;
  #pragma unroll
  for (int e = 0; e < E_; ++e) if (e != i1 && acc[e] > l2) { l2 = acc[e]; i2 = e; }
  float e2 = __expf(l2 - l1);
  float w1 = 1.f / (1.f + e2);
  float w2 = e2 / (1.f + e2);
  #pragma unroll
  for (int e = 0; e < E_; ++e)
    gates[(long)t * E_ + e] = (e == i1) ? w1 : ((e == i2) ? w2 : 0.f);
}

// ---------------- copy f32 (seed output with residual h) ----------------
__global__ void copy_f32(const float* __restrict__ in, float* __restrict__ out, long n) {
  long i = (long)blockIdx.x * blockDim.x + threadIdx.x;
  long stride = (long)gridDim.x * blockDim.x;
  for (; i < n; i += stride) out[i] = in[i];
}

// ---------------- launch ----------------
extern "C" void kernel_launch(void* const* d_in, const int* in_sizes, int n_in,
                              void* d_out, int out_size, void* d_ws, size_t ws_size,
                              hipStream_t stream) {
  const float* x        = (const float*)d_in[0];
  const float* norm1_w  = (const float*)d_in[1];
  const float* norm2_w  = (const float*)d_in[2];
  const float* Wq       = (const float*)d_in[3];
  const float* Wk       = (const float*)d_in[4];
  const float* Wv       = (const float*)d_in[5];
  const float* Wo       = (const float*)d_in[6];
  const float* router_w = (const float*)d_in[7];
  const float* router_b = (const float*)d_in[8];
  const float* W1       = (const float*)d_in[9];
  const float* b1       = (const float*)d_in[10];
  const float* W2       = (const float*)d_in[11];
  const float* b2       = (const float*)d_in[12];
  const float* W3       = (const float*)d_in[13];
  const float* b3       = (const float*)d_in[14];
  float* out            = (float*)d_out;

  // carve workspace (256B aligned chunks)
  size_t off = 0;
  auto carve = [&](size_t bytes) -> void* {
    void* p = (char*)d_ws + off;
    off += (bytes + 255) & ~(size_t)255;
    return p;
  };
  bf16* xr    = (bf16*)carve((size_t)BS_ * D_ * 2);
  bf16* qb    = (bf16*)carve((size_t)BS_ * D_ * 2);
  bf16* kb    = (bf16*)carve((size_t)BS_ * G_ * DK_ * 2);
  bf16* vtb   = (bf16*)carve((size_t)BS_ * G_ * DK_ * 2);   // transposed V
  bf16* attnb = (bf16*)carve((size_t)BS_ * D_ * 2);
  float* hbuf = (float*)carve((size_t)BS_ * D_ * 4);
  bf16* ynb   = (bf16*)carve((size_t)BS_ * D_ * 2);
  float* gates= (float*)carve((size_t)BS_ * E_ * 4);
  bf16* h1    = (bf16*)carve((size_t)BS_ * FF_ * 2);
  bf16* WqT   = (bf16*)carve((size_t)D_ * D_ * 2);
  bf16* WkT   = (bf16*)carve((size_t)D_ * G_ * DK_ * 2);
  bf16* WvT   = (bf16*)carve((size_t)D_ * G_ * DK_ * 2);
  bf16* WoT   = (bf16*)carve((size_t)D_ * D_ * 2);
  bf16* W1T   = (bf16*)carve((size_t)E_ * D_ * FF_ * 2);
  bf16* W2T   = (bf16*)carve((size_t)E_ * D_ * FF_ * 2);
  bf16* W3T   = (bf16*)carve((size_t)E_ * FF_ * D_ * 2);

  const dim3 tb(32, 8);
  // 1) fp32 -> bf16 with transpose (weights become N-major, K-contiguous)
  cvt_transpose<<<dim3(D_ / 32, D_ / 32, 1), tb, 0, stream>>>(Wq, WqT, D_, D_, 0, 0);
  cvt_transpose<<<dim3((G_ * DK_) / 32, D_ / 32, 1), tb, 0, stream>>>(Wk, WkT, D_, G_ * DK_, 0, 0);
  cvt_transpose<<<dim3((G_ * DK_) / 32, D_ / 32, 1), tb, 0, stream>>>(Wv, WvT, D_, G_ * DK_, 0, 0);
  cvt_transpose<<<dim3(D_ / 32, D_ / 32, 1), tb, 0, stream>>>(Wo, WoT, D_, D_, 0, 0);
  cvt_transpose<<<dim3(FF_ / 32, D_ / 32, E_), tb, 0, stream>>>(
      W1, W1T, D_, FF_, (long)D_ * FF_, (long)D_ * FF_);
  cvt_transpose<<<dim3(FF_ / 32, D_ / 32, E_), tb, 0, stream>>>(
      W2, W2T, D_, FF_, (long)D_ * FF_, (long)D_ * FF_);
  cvt_transpose<<<dim3(D_ / 32, FF_ / 32, E_), tb, 0, stream>>>(
      W3, W3T, FF_, D_, (long)FF_ * D_, (long)FF_ * D_);

  // 2) rmsnorm1 + rope -> xr (bf16)
  rmsnorm_rope_kernel<<<BS_, 256, 0, stream>>>(x, norm1_w, xr);

  // 3) Q/K/V projections (WMMA); V written pre-transposed for attention
  gemm_bf16_out<<<dim3(D_ / 128, BS_ / 64), 128, 0, stream>>>(xr, WqT, qb, D_, D_);
  gemm_bf16_out<<<dim3((G_ * DK_) / 128, BS_ / 64), 128, 0, stream>>>(xr, WkT, kb, G_ * DK_, D_);
  gemm_bf16_outT<<<dim3((G_ * DK_) / 128, BS_ / 64), 128, 0, stream>>>(xr, WvT, vtb, G_ * DK_, D_);

  // 4) causal GQA flash attention (WMMA, double-buffered async K/V staging)
  attn_kernel<<<dim3(S_ / 64, NH_, B_), 128, 0, stream>>>(qb, kb, vtb, attnb);

  // 5) output projection + residual -> h (f32)
  gemm_wo_resid<<<dim3(D_ / 128, BS_ / 64), 128, 0, stream>>>(attnb, WoT, x, hbuf, D_, D_);

  // 6) rmsnorm2 -> yn (bf16)
  rmsnorm_kernel<<<BS_, 256, 0, stream>>>(hbuf, norm2_w, ynb);

  // 7) router logits + top-2 gates
  router_gates<<<(BS_ + 255) / 256, 256, 0, stream>>>(ynb, router_w, router_b, gates);

  // 8) seed output with residual h
  copy_f32<<<2048, 256, 0, stream>>>(hbuf, out, (long)BS_ * D_);

  // 9) MoE experts: fused dual GEMM (silu * gate-proj), then gated W3 accumulate
  for (int e = 0; e < E_; ++e) {
    const bf16* W1e = W1T + (long)e * D_ * FF_;
    const bf16* W2e = W2T + (long)e * D_ * FF_;
    const bf16* W3e = W3T + (long)e * FF_ * D_;
    gemm_dual_silu<<<dim3(FF_ / 64, BS_ / 64), 128, 0, stream>>>(
        ynb, W1e, W2e, b1 + (long)e * FF_, b2 + (long)e * FF_, h1, FF_, D_);
    gemm_w3_gate<<<dim3(D_ / 128, BS_ / 64), 128, 0, stream>>>(
        h1, W3e, b3 + (long)e * D_, gates + e, out, D_, FF_);
  }
}
